// XLNetRelativeAttention_55336358642049
// MI455X (gfx1250) — matrix operand order repaired and checked by reading, not compiled, not run on hardware
//
#include <hip/hip_runtime.h>

#define QLEN 1024
#define BSZ  4
#define DM   1024
#define NH   16
#define DH   64
#define RLEN 2048

typedef __attribute__((ext_vector_type(16))) __bf16 v16bf;
typedef __attribute__((ext_vector_type(8)))  float  v8f;

static __device__ __forceinline__ __bf16 f2bf(float f) { return (__bf16)f; }
static __device__ __forceinline__ float  bf2f(__bf16 b) { return (float)b; }

static __device__ __forceinline__ v8f wmma_bf16(v16bf a, v16bf b, v8f c) {
  // D = A(16x32 bf16) x B(32x16 bf16) + C(16x16 f32)
  return __builtin_amdgcn_wmma_f32_16x16x32_bf16(false, a, false, b, (short)0, c,
                                                 false, false);
}
static __device__ __forceinline__ v8f zero8() {
  v8f z = {0.f, 0.f, 0.f, 0.f, 0.f, 0.f, 0.f, 0.f};
  return z;
}
// A-fragment swizzle within a 32-element K-chunk: lane(lmod,lhalf) wants
// elems K = lhalf*8 + {0..7, 16..23}. Storing chunk element dd at position
// fragswz(dd) makes each lane's 16 elements contiguous at offset lhalf*16.
static __device__ __forceinline__ int fragswz(int dd) {
  const int blk = dd >> 3;
  return (blk == 1 || blk == 2) ? (dd ^ 24) : dd;
}

// ---------------------------------------------------------------------------
// Projection GEMM: C[M, NH*DH] = A[M, DM](f32->bf16) x W[DM, NH*DH](f32->bf16)
// Each wave computes a 16(M) x 64(N=one head) tile with 4 WMMA accumulators.
// MODE 0: Q   -> q+rw, q+rr, q+rs (all linear)  [b][n][i][d]
// MODE 1: K_h -> bf16 [b][n][j][d]
// MODE 2: V   -> bf16 transposed [b][n][d][j]
// MODE 3: K_r -> bf16 [b][n][j'][d]   (rows = RLEN*BSZ)
// ---------------------------------------------------------------------------
template <int MODE>
__global__ __launch_bounds__(256) void proj_kernel(
    const float* __restrict__ A, const float* __restrict__ W,
    __bf16* __restrict__ out0, __bf16* __restrict__ out1, __bf16* __restrict__ out2,
    const float* __restrict__ b0, const float* __restrict__ b1,
    const float* __restrict__ b2)
{
  const int w = threadIdx.x >> 5, lane = threadIdx.x & 31;
  const int lmod = lane & 15, lhalf = lane >> 4;
  const int n  = blockIdx.y;
  const int i0 = (blockIdx.x * 8 + w) * 16;
  const int N0 = n * DH;

  v8f acc[4];
#pragma unroll
  for (int t = 0; t < 4; t++) acc[t] = zero8();

  const int arow = i0 + lmod;
#pragma unroll 2
  for (int kk = 0; kk < DM; kk += 32) {
    // A fragment: lane holds row M=lmod; elems 0-7 -> K=kk+lhalf*8+e,
    // elems 8-15 -> K=kk+lhalf*8+16+e (ISA 16-bit A layout)
    v16bf af;
    const float* ap = A + (size_t)arow * DM + kk + lhalf * 8;
#pragma unroll
    for (int e = 0; e < 8; e++) { af[e] = f2bf(ap[e]); af[8 + e] = f2bf(ap[16 + e]); }
#pragma unroll
    for (int t = 0; t < 4; t++) {
      const int col = N0 + t * 16 + lmod;                 // B: lane holds N=col
      const float* wp = W + (size_t)(kk + 8 * lhalf) * (NH * DH) + col;
      v16bf bfv;                                          // elems -> K=8*lhalf+(e<8?e:e+8)
#pragma unroll
      for (int e = 0; e < 16; e++) bfv[e] = f2bf(wp[(size_t)(e < 8 ? e : e + 8) * (NH * DH)]);
      acc[t] = wmma_bf16(af, bfv, acc[t]);
    }
  }

  // C layout: lanes 0-15 -> rows v (0..7), lanes 16-31 -> rows v+8; N = lmod
#pragma unroll
  for (int t = 0; t < 4; t++) {
#pragma unroll
    for (int v = 0; v < 8; v++) {
      const int r    = v + 8 * lhalf;
      const int grow = i0 + r;                    // A row = seq*BSZ + b
      const int d    = t * 16 + lmod;
      const int c    = N0 + d;
      const int bi   = grow & (BSZ - 1);
      const int seq  = grow >> 2;
      const int bn   = bi * NH + n;
      const float val = acc[t][v];
      if (MODE == 0) {
        // linear d for qw/qr (contiguous-K v16bf A-frag loads downstream)
        const int dswz = ((d >> 5) << 5) + fragswz(d & 31);
        size_t base = ((size_t)bn * QLEN + seq) * DH;
        out0[base + d] = f2bf(val + b0[c]);
        out1[base + d] = f2bf(val + b1[c]);
        out2[base + d]    = f2bf(val + b2[c]);   // qs stays linear (ef kernel)
      } else if (MODE == 1) {
        out0[((size_t)bn * QLEN + seq) * DH + d] = f2bf(val);
      } else if (MODE == 2) {
        out0[((size_t)bn * DH + d) * QLEN + seq] = f2bf(val);
      } else {
        out0[((size_t)bn * RLEN + seq) * DH + d] = f2bf(val);
      }
    }
  }
}

// ---------------------------------------------------------------------------
// ef[b][n][i][s] = sum_d (q+rs)[i,b,n,d] * seg_embed[s,n,d]
// ---------------------------------------------------------------------------
__global__ __launch_bounds__(256) void ef_kernel(
    const __bf16* __restrict__ qs, const float* __restrict__ seg_embed,
    float* __restrict__ ef)
{
  int flat = blockIdx.x * blockDim.x + threadIdx.x;  // [b][n][i]
  if (flat >= BSZ * NH * QLEN) return;
  int i = flat % QLEN;
  int n = (flat / QLEN) % NH;
  int b = flat / (QLEN * NH);
  int bn = b * NH + n;
  const __bf16* qp = qs + ((size_t)bn * QLEN + i) * DH;
  const float* e0 = seg_embed + (size_t)n * DH;
  const float* e1 = seg_embed + (size_t)(NH + n) * DH;
  float s0 = 0.f, s1 = 0.f;
#pragma unroll 8
  for (int d = 0; d < DH; d++) {
    float q = bf2f(qp[d]);
    s0 += q * e0[d];
    s1 += q * e1[d];
  }
  ef[((size_t)bn * QLEN + i) * 2 + 0] = s0;
  ef[((size_t)bn * QLEN + i) * 2 + 1] = s1;
}

// ---------------------------------------------------------------------------
// Fused flash attention with LDS-staged rel_shift:
//   bd[i,j] = raw[i, j - i + 1024]  (always in [1,2047] for this shape)
// Per wave: 16 query rows; stream j in tiles of 32; online softmax; P·V in WMMA.
// qw/qr and the LDS P tile are stored linear -> all A-frags are single
// contiguous 32B loads; all B-frags (kh/kr/vt) are single contiguous 32B loads.
// ---------------------------------------------------------------------------
#define FWAVES 4
__global__ __launch_bounds__(128) void flash_kernel(
    const __bf16* __restrict__ qw, const __bf16* __restrict__ qr,
    const __bf16* __restrict__ kh, const __bf16* __restrict__ kr,
    const __bf16* __restrict__ vt, const float* __restrict__ ef,
    const float* __restrict__ seg_mat, const float* __restrict__ attn_mask,
    __bf16* __restrict__ av)
{
  __shared__ float  stage[FWAVES][16][48];   // raw-bd band, then reused for S
  __shared__ __bf16 pbuf[FWAVES][16][32];    // P tile (bf16, linear)
  __shared__ float  rowm[FWAVES][16], rowl[FWAVES][16], rowc[FWAVES][16];
  __shared__ float  pmax[FWAVES][16][2], psum[FWAVES][16][2];
  __shared__ float  efs[FWAVES][16][2];

  const int w = threadIdx.x >> 5, lane = threadIdx.x & 31;
  const int lmod = lane & 15, lhalf = lane >> 4;
  const int n = blockIdx.y, b = blockIdx.z;
  const int i0 = (blockIdx.x * FWAVES + w) * 16;
  const int bn = b * NH + n;
  const float scale = 0.125f;  // 1/sqrt(64)

  // Q fragments (linear storage): chunk kk_idx at +32*kk_idx, lane half
  // at +16 -> contiguous 32B per fragment.
  const __bf16* qwl = qw + ((size_t)bn * QLEN + i0 + lmod) * DH + lhalf * 16;
  const __bf16* qrl = qr + ((size_t)bn * QLEN + i0 + lmod) * DH + lhalf * 16;
  const v16bf aqw0 = *(const v16bf*)(qwl);
  const v16bf aqw1 = *(const v16bf*)(qwl + 32);
  const v16bf aqr0 = *(const v16bf*)(qrl);
  const v16bf aqr1 = *(const v16bf*)(qrl + 32);

  if (lane < 16) {
    efs[w][lane][0] = ef[((size_t)bn * QLEN + i0 + lane) * 2 + 0];
    efs[w][lane][1] = ef[((size_t)bn * QLEN + i0 + lane) * 2 + 1];
    rowm[w][lane] = -1e30f;
    rowl[w][lane] = 0.f;
  }

  v8f accO[4];
#pragma unroll
  for (int t = 0; t < 4; t++) accO[t] = zero8();

  for (int j0 = 0; j0 < QLEN; j0 += 32) {
    if (j0 + 32 < QLEN)
      __builtin_prefetch(kh + ((size_t)bn * QLEN + j0 + 32 + lane) * DH, 0, 0);

    // ---- content scores ac : two 16x16 tiles -------------------------------
    v8f s0 = zero8(), s1 = zero8();
    {
      const __bf16* kp0 = kh + ((size_t)bn * QLEN + j0 + lmod) * DH + 16 * lhalf;
      const __bf16* kp1 = kp0 + (size_t)16 * DH;
      s0 = wmma_bf16(aqw0, *(const v16bf*)(kp0),      s0);
      s0 = wmma_bf16(aqw1, *(const v16bf*)(kp0 + 32), s0);
      s1 = wmma_bf16(aqw0, *(const v16bf*)(kp1),      s1);
      s1 = wmma_bf16(aqw1, *(const v16bf*)(kp1 + 32), s1);
    }

    // ---- position scores: raw band of 3 tiles, staged, then shifted --------
    const int base0 = j0 - i0 + 1024;   // raw column for (r=0,c=0)
#pragma unroll
    for (int t = 0; t < 3; t++) {
      v8f rb = zero8();
      const int jr = base0 - 16 + t * 16 + lmod;   // always in [0, 2048)
      const __bf16* kp = kr + ((size_t)bn * RLEN + jr) * DH + 16 * lhalf;
      rb = wmma_bf16(aqr0, *(const v16bf*)(kp),      rb);
      rb = wmma_bf16(aqr1, *(const v16bf*)(kp + 32), rb);
#pragma unroll
      for (int v = 0; v < 8; v++)
        stage[w][v + 8 * lhalf][t * 16 + lmod] = rb[v];
    }

    // ---- combine ac + shifted bd + ef + mask, scale ------------------------
#pragma unroll
    for (int jt = 0; jt < 2; jt++) {
      v8f& s = jt ? s1 : s0;
      const int c_ = jt * 16 + lmod;
      const int j  = j0 + c_;
#pragma unroll
      for (int v = 0; v < 8; v++) {
        const int rr = v + 8 * lhalf;
        const float raw = stage[w][rr][c_ - rr + 16];   // rel_shift read
        const int i = i0 + rr;
        const float e0 = efs[w][rr][0], e1 = efs[w][rr][1];
        const float diff = seg_mat[(((size_t)i * QLEN + j) * BSZ + b) * 2 + 1];
        const float msk  = attn_mask[((size_t)i * QLEN + j) * BSZ + b];
        s[v] = (s[v] + raw + e0 + (e1 - e0) * diff) * scale - 1e30f * msk;
      }
    }

    // ---- write S tile to LDS (reuse stage cols 0..31) ----------------------
#pragma unroll
    for (int jt = 0; jt < 2; jt++) {
      v8f& s = jt ? s1 : s0;
#pragma unroll
      for (int v = 0; v < 8; v++)
        stage[w][v + 8 * lhalf][jt * 16 + lmod] = s[v];
    }

    // ---- online softmax: lane owns (row=lmod, half=lhalf) ------------------
    float lm = -3.0e38f;
#pragma unroll
    for (int c = 0; c < 16; c++)
      lm = fmaxf(lm, stage[w][lmod][lhalf * 16 + c]);
    pmax[w][lmod][lhalf] = lm;
    const float tmax  = fmaxf(pmax[w][lmod][0], pmax[w][lmod][1]);
    const float m_old = rowm[w][lmod];
    const float m_new = fmaxf(m_old, tmax);
    const float corr  = __expf(m_old - m_new);
    float ls = 0.f;
#pragma unroll
    for (int c = 0; c < 16; c++) {
      const int cg = lhalf * 16 + c;
      float p = __expf(stage[w][lmod][cg] - m_new);
      pbuf[w][lmod][cg] = f2bf(p);   // store P linear (contiguous-K A-frag)
      ls += p;
    }
    psum[w][lmod][lhalf] = ls;
    if (lhalf == 0) {
      rowl[w][lmod] = rowl[w][lmod] * corr + psum[w][lmod][0] + psum[w][lmod][1];
      rowm[w][lmod] = m_new;
      rowc[w][lmod] = corr;
    }

    // ---- O = O*corr + P x V ------------------------------------------------
    {
      float cc[8];
#pragma unroll
      for (int v = 0; v < 8; v++) cc[v] = rowc[w][v + 8 * lhalf];
#pragma unroll
      for (int t = 0; t < 4; t++)
#pragma unroll
        for (int v = 0; v < 8; v++) accO[t][v] *= cc[v];

      // A-frag: single contiguous 32B LDS read (linear P)
      const v16bf ap = *(const v16bf*)(&pbuf[w][lmod][lhalf * 16]);
#pragma unroll
      for (int t = 0; t < 4; t++) {
        const int d = t * 16 + lmod;
        const __bf16* vp = vt + ((size_t)bn * DH + d) * QLEN + j0 + 16 * lhalf;
        accO[t] = wmma_bf16(ap, *(const v16bf*)vp, accO[t]);
      }
    }
  }

  // ---- normalize, store attn_vec bf16 rows [(i,b), (n,d)] frag-swizzled ----
#pragma unroll
  for (int t = 0; t < 4; t++) {
#pragma unroll
    for (int v = 0; v < 8; v++) {
      const int rr = v + 8 * lhalf;
      const int i  = i0 + rr;
      const int d  = t * 16 + lmod;
      const int dswz = ((d >> 5) << 5) + fragswz(d & 31);
      const float val = accO[t][v] / rowl[w][rr];
      av[((size_t)(i * BSZ + b)) * DM + n * DH + dswz] = f2bf(val);
    }
  }
}

// ---------------------------------------------------------------------------
// Output projection: y[(i,b), h'] = sum_c av[(i,b), c]*proj_o[h', c] + h
// av is frag-swizzled per 32-chunk, so the A-frag is one contiguous 32B load.
// The weight B-frag needs the matching K permutation: within chunk kk, lane
// elems e map to K = kk + lhalf*8 + (e<8 ? e : e+8)  (same as ISA A layout).
// ---------------------------------------------------------------------------
__global__ __launch_bounds__(256) void outproj_kernel(
    const __bf16* __restrict__ av, const float* __restrict__ Wo,
    const float* __restrict__ hres, float* __restrict__ y)
{
  const int w = threadIdx.x >> 5, lane = threadIdx.x & 31;
  const int lmod = lane & 15, lhalf = lane >> 4;
  const int i0 = (blockIdx.x * 8 + w) * 16;
  const int N0 = blockIdx.y * 64;

  v8f acc[4];
#pragma unroll
  for (int t = 0; t < 4; t++) acc[t] = zero8();

  const __bf16* arow = av + (size_t)(i0 + lmod) * DM;
#pragma unroll 2
  for (int kk = 0; kk < DM; kk += 32) {
    const v16bf af = *(const v16bf*)(arow + kk + lhalf * 16);
#pragma unroll
    for (int t = 0; t < 4; t++) {
      const int col = N0 + t * 16 + lmod;                  // output channel h'
      const float* wp = Wo + (size_t)col * DM + kk + 8 * lhalf;
      v16bf bfv;
#pragma unroll
      for (int e = 0; e < 16; e++) bfv[e] = f2bf(wp[e < 8 ? e : e + 8]);
      acc[t] = wmma_bf16(af, bfv, acc[t]);
    }
  }
#pragma unroll
  for (int t = 0; t < 4; t++) {
#pragma unroll
    for (int v = 0; v < 8; v++) {
      const int r = v + 8 * lhalf;
      const int grow = i0 + r;
      const int c = N0 + t * 16 + lmod;
      y[(size_t)grow * DM + c] = acc[t][v] + hres[(size_t)grow * DM + c];
    }
  }
}

// ---------------------------------------------------------------------------
// LayerNorm over DM per (i,b) row.
// ---------------------------------------------------------------------------
__global__ __launch_bounds__(256) void ln_kernel(
    const float* __restrict__ y, const float* __restrict__ gamma,
    const float* __restrict__ beta, float* __restrict__ out)
{
  __shared__ float red[256];
  const int row = blockIdx.x;
  const float* yr = y + (size_t)row * DM;

  float s = 0.f;
  for (int c = threadIdx.x; c < DM; c += 256) s += yr[c];
  red[threadIdx.x] = s; __syncthreads();
  for (int st = 128; st > 0; st >>= 1) {
    if (threadIdx.x < st) red[threadIdx.x] += red[threadIdx.x + st];
    __syncthreads();
  }
  const float mu = red[0] * (1.0f / DM);
  __syncthreads();

  float v = 0.f;
  for (int c = threadIdx.x; c < DM; c += 256) { float d = yr[c] - mu; v += d * d; }
  red[threadIdx.x] = v; __syncthreads();
  for (int st = 128; st > 0; st >>= 1) {
    if (threadIdx.x < st) red[threadIdx.x] += red[threadIdx.x + st];
    __syncthreads();
  }
  const float rstd = rsqrtf(red[0] * (1.0f / DM) + 1e-12f);

  for (int c = threadIdx.x; c < DM; c += 256)
    out[(size_t)row * DM + c] = (yr[c] - mu) * rstd * gamma[c] + beta[c];
}

// ---------------------------------------------------------------------------
extern "C" void kernel_launch(void* const* d_in, const int* in_sizes, int n_in,
                              void* d_out, int out_size, void* d_ws, size_t ws_size,
                              hipStream_t stream)
{
  const float* h         = (const float*)d_in[0];
  const float* r         = (const float*)d_in[1];
  const float* attn_mask = (const float*)d_in[2];
  const float* seg_mat   = (const float*)d_in[3];
  const float* proj_q    = (const float*)d_in[4];
  const float* proj_k    = (const float*)d_in[5];
  const float* proj_v    = (const float*)d_in[6];
  const float* proj_o    = (const float*)d_in[7];
  const float* proj_r    = (const float*)d_in[8];
  const float* r_w_bias  = (const float*)d_in[9];
  const float* r_r_bias  = (const float*)d_in[10];
  const float* r_s_bias  = (const float*)d_in[11];
  const float* seg_embed = (const float*)d_in[12];
  const float* ln_gamma  = (const float*)d_in[13];
  const float* ln_beta   = (const float*)d_in[14];
  float* out = (float*)d_out;

  // workspace layout (all sizes power-of-two multiples; base is aligned)
  const size_t SZ_Q = (size_t)BSZ * NH * QLEN * DH;      // 4,194,304 elems
  __bf16* qw  = (__bf16*)d_ws;
  __bf16* qr  = qw + SZ_Q;
  __bf16* qs  = qr + SZ_Q;
  __bf16* kh  = qs + SZ_Q;
  __bf16* vt  = kh + SZ_Q;
  __bf16* kr  = vt + SZ_Q;                               // 2*SZ_Q elems
  float*  efb = (float*)(kr + 2 * SZ_Q);                 // BSZ*NH*QLEN*2 f32
  __bf16* av  = (__bf16*)(efb + (size_t)BSZ * NH * QLEN * 2);
  float*  y   = (float*)(av + (size_t)QLEN * BSZ * DM);

  proj_kernel<0><<<dim3(32, NH), 256, 0, stream>>>(h, proj_q, qw, qr, qs,
                                                   r_w_bias, r_r_bias, r_s_bias);
  proj_kernel<1><<<dim3(32, NH), 256, 0, stream>>>(h, proj_k, kh, nullptr, nullptr,
                                                   nullptr, nullptr, nullptr);
  proj_kernel<2><<<dim3(32, NH), 256, 0, stream>>>(h, proj_v, vt, nullptr, nullptr,
                                                   nullptr, nullptr, nullptr);
  proj_kernel<3><<<dim3(64, NH), 256, 0, stream>>>(r, proj_r, kr, nullptr, nullptr,
                                                   nullptr, nullptr, nullptr);
  ef_kernel<<<dim3((BSZ * NH * QLEN) / 256), 256, 0, stream>>>(qs, seg_embed, efb);
  flash_kernel<<<dim3(QLEN / (16 * FWAVES), NH, BSZ), 32 * FWAVES, 0, stream>>>(
      qw, qr, kh, kr, vt, efb, seg_mat, attn_mask, av);
  outproj_kernel<<<dim3(32, DM / 64), 256, 0, stream>>>(av, proj_o, h, y);
  ln_kernel<<<dim3(QLEN * BSZ), 256, 0, stream>>>(y, ln_gamma, ln_beta, out);

  (void)in_sizes; (void)n_in; (void)out_size; (void)ws_size;
}
